// LocalConceptTransforming_87497073754450
// MI455X (gfx1250) — matrix operand order repaired and checked by reading, not compiled, not run on hardware
//
#include <hip/hip_runtime.h>
#include <hip/hip_bf16.h>
#include <math.h>

// ---------------------------------------------------------------------------
// CLIP-style transformer encoder for MI455X (gfx1250), f16 WMMA + f32 accum.
// GEMMs: 128x128 block tiles, 32x64 wave tiles, double-buffered LDS filled
// with GLOBAL_LOAD_ASYNC_TO_LDS_B128 (ASYNCcnt), v_wmma_f32_16x16x32_f16.
// ---------------------------------------------------------------------------

typedef __attribute__((ext_vector_type(16))) _Float16 v16h;
typedef __attribute__((ext_vector_type(8)))  _Float16 v8h;   // 16B = one b128
typedef __attribute__((ext_vector_type(8)))  float    v8f;

#define S_    64
#define B_    1024
#define D_    512
#define H_    8
#define DH_   64
#define NL_   4
#define KBITS 32
#define DFF_  2048
#define NTOK  (S_ * B_)          // 65536 token rows

// ---- WMMA helpers ---------------------------------------------------------

__device__ __forceinline__ v8f wmma16x16x32(v16h a, v16h b, v8f c) {
  return __builtin_amdgcn_wmma_f32_16x16x32_f16(
      /*neg_a=*/false, a, /*neg_b=*/false, b,
      /*c_mod=*/(short)0, c, /*reuse_a=*/false, /*reuse_b=*/false);
}

// A fragment (16x32, f16). Lanes 0-15: row M=lane, K={0..7,16..23};
// lanes 16-31: row M=lane-16, K={8..15,24..31}. stride in halves.
__device__ __forceinline__ v16h load_frag_a(const _Float16* base, int stride) {
  const int lane = threadIdx.x & 31;
  const int l16  = lane & 15;
  const int lh   = lane >> 4;
  const _Float16* p = base + l16 * stride + lh * 8;
  union { v16h v; v8h h[2]; } u;
  u.h[0] = *(const v8h*)(p);
  u.h[1] = *(const v8h*)(p + 16);
  return u.v;
}

// B fragment (32x16, f16), tile stored as Wt[n][k] (row-major weights == B^T).
// Lane l16 = column n; lanes 0-15: K={0..7,16..23}, lanes 16-31: K={8..15,24..31}.
__device__ __forceinline__ v16h load_frag_b(const _Float16* base, int stride) {
  const int lane = threadIdx.x & 31;
  const int l16  = lane & 15;
  const int lh   = lane >> 4;
  const _Float16* p = base + l16 * stride + lh * 8;
  union { v16h v; v8h h[2]; } u;
  u.h[0] = *(const v8h*)(p);
  u.h[1] = *(const v8h*)(p + 16);
  return u.v;
}

// ---- CDNA5 async global->LDS copy (ASYNCcnt) ------------------------------
// Generic-pointer low 32 bits are the wave-relative LDS byte address
// INST_OFFSET is added to both
// the LDS and global addresses, so one base covers consecutive 16B chunks.

__device__ __forceinline__ void async_ld_lds_b128x2(void* lds, const void* g) {
  unsigned l = (unsigned)(size_t)lds;
  asm volatile("global_load_async_to_lds_b128 %0, %1, off"
               :: "v"(l), "v"(g) : "memory");
  asm volatile("global_load_async_to_lds_b128 %0, %1, off offset:16"
               :: "v"(l), "v"(g) : "memory");
}

__device__ __forceinline__ void wait_async0() {
  asm volatile("s_wait_asynccnt 0x0" ::: "memory");
}

// ---- elementwise kernels --------------------------------------------------

__global__ __launch_bounds__(256) void pe_add_kernel(
    const float* __restrict__ xin, float* __restrict__ X) {
  size_t idx = (size_t)blockIdx.x * 256 + threadIdx.x;   // over S*B*D
  int d = (int)(idx & (D_ - 1));
  int s = (int)(idx / ((size_t)B_ * D_));
  int i2 = (d >> 1) * 2;
  float div = __expf(-(9.210340371976184f / (float)D_) * (float)i2);
  float ang = (float)s * div;
  float pe  = (d & 1) ? __cosf(ang) : __sinf(ang);
  X[idx] = xin[idx] + pe * 0.04419417382415922f;   // 1/sqrt(512)
}

__global__ __launch_bounds__(256) void f2h_kernel(
    const float* __restrict__ src, _Float16* __restrict__ dst) {
  size_t idx = (size_t)blockIdx.x * 256 + threadIdx.x;
  dst[idx] = (_Float16)src[idx];
}

// LayerNorm over D=512, one wave per row (32 lanes x 16 elems), f16 output.
__global__ __launch_bounds__(256) void ln_kernel(
    const float* __restrict__ X, const float* __restrict__ g,
    const float* __restrict__ b, _Float16* __restrict__ Hout) {
  const int wave = threadIdx.x >> 5;
  const int lane = threadIdx.x & 31;
  const int row  = blockIdx.x * 8 + wave;
  const float* xr = X + (size_t)row * D_;
  float v[16];
  float s = 0.0f;
#pragma unroll
  for (int i = 0; i < 16; i++) { v[i] = xr[lane + 32 * i]; s += v[i]; }
#pragma unroll
  for (int off = 16; off >= 1; off >>= 1) s += __shfl_xor(s, off, 32);
  const float mean = s * (1.0f / D_);
  float vs = 0.0f;
#pragma unroll
  for (int i = 0; i < 16; i++) { float d = v[i] - mean; vs += d * d; }
#pragma unroll
  for (int off = 16; off >= 1; off >>= 1) vs += __shfl_xor(vs, off, 32);
  const float inv = rsqrtf(vs * (1.0f / D_) + 1e-5f);
  _Float16* hr = Hout + (size_t)row * D_;
#pragma unroll
  for (int i = 0; i < 16; i++) {
    int c = lane + 32 * i;
    hr[c] = (_Float16)((v[i] - mean) * inv * g[c] + b[c]);
  }
}

// ---- GEMM: out = A[N,K](f16) @ W[M,K]^T(f16) + bias, epilogue by mode -----
// mode 0: store f16 (QKV)   mode 1: outF += val (residual, f32)
// mode 2: QuickGELU -> f16 (MLP up-proj)

#define BM 128
#define BN 128
#define BK 32
#define LDS_AS 40
#define LDS_BS 40

__device__ __forceinline__ void store_tile(
    v8f acc, int rowB, int col, int M, const float* __restrict__ bias,
    _Float16* __restrict__ outH, float* __restrict__ outF, int mode) {
  const float bv = bias[col];
#pragma unroll
  for (int v = 0; v < 8; v++) {
    size_t idx = (size_t)(rowB + v) * M + col;
    float val = acc[v] + bv;
    if (mode == 0) {
      outH[idx] = (_Float16)val;
    } else if (mode == 1) {
      outF[idx] += val;
    } else {
      float gg = val / (1.0f + __expf(-1.702f * val));
      outH[idx] = (_Float16)gg;
    }
  }
}

__global__ __launch_bounds__(256) void gemm_f16_kernel(
    const _Float16* __restrict__ A, const _Float16* __restrict__ W,
    const float* __restrict__ bias, _Float16* __restrict__ outH,
    float* __restrict__ outF, int M, int K, int mode) {
  __shared__ __align__(16) _Float16 As[2][BM * LDS_AS];
  __shared__ __align__(16) _Float16 Bs[2][BN * LDS_BS];

  const int tid  = threadIdx.x;
  const int lane = tid & 31;
  const int wave = tid >> 5;
  const int l16  = lane & 15;
  const int lh   = lane >> 4;

  const int rowBlock = blockIdx.y * BM;
  const int colBlock = blockIdx.x * BN;
  const int wRow = (wave >> 1) * 32;   // 4 wave-rows x 32
  const int wCol = (wave & 1) * 64;    // 2 wave-cols x 64

  v8f acc[2][4];
#pragma unroll
  for (int i = 0; i < 2; i++)
#pragma unroll
    for (int j = 0; j < 4; j++) acc[i][j] = (v8f){};

  // per-thread tile-fill assignment: 32B (16 halves) per tile
  const int tr  = tid >> 1;            // 0..127
  const int tkk = (tid & 1) * 16;

  const _Float16* gA = A + (size_t)(rowBlock + tr) * K + tkk;
  const _Float16* gB = W + (size_t)(colBlock + tr) * K + tkk;
  const int ldsoff = tr * LDS_AS + tkk;

  const int nk = K / BK;
  // prologue: fill buffer 0
  async_ld_lds_b128x2(&As[0][ldsoff], gA);
  async_ld_lds_b128x2(&Bs[0][ldsoff], gB);
  wait_async0();
  __syncthreads();

  for (int kt = 0; kt < nk; kt++) {
    const int cur = kt & 1;
    if (kt + 1 < nk) {   // prefetch next tile into the other buffer
      const int knext = (kt + 1) * BK;
      async_ld_lds_b128x2(&As[cur ^ 1][ldsoff], gA + knext);
      async_ld_lds_b128x2(&Bs[cur ^ 1][ldsoff], gB + knext);
    }

    v16h a0 = load_frag_a(&As[cur][(wRow +  0) * LDS_AS], LDS_AS);
    v16h a1 = load_frag_a(&As[cur][(wRow + 16) * LDS_AS], LDS_AS);
    v16h b0 = load_frag_b(&Bs[cur][(wCol +  0) * LDS_BS], LDS_BS);
    v16h b1 = load_frag_b(&Bs[cur][(wCol + 16) * LDS_BS], LDS_BS);
    v16h b2 = load_frag_b(&Bs[cur][(wCol + 32) * LDS_BS], LDS_BS);
    v16h b3 = load_frag_b(&Bs[cur][(wCol + 48) * LDS_BS], LDS_BS);

    acc[0][0] = wmma16x16x32(a0, b0, acc[0][0]);
    acc[0][1] = wmma16x16x32(a0, b1, acc[0][1]);
    acc[0][2] = wmma16x16x32(a0, b2, acc[0][2]);
    acc[0][3] = wmma16x16x32(a0, b3, acc[0][3]);
    acc[1][0] = wmma16x16x32(a1, b0, acc[1][0]);
    acc[1][1] = wmma16x16x32(a1, b1, acc[1][1]);
    acc[1][2] = wmma16x16x32(a1, b2, acc[1][2]);
    acc[1][3] = wmma16x16x32(a1, b3, acc[1][3]);

    wait_async0();        // next-tile fills complete (no-op on last iter)
    __syncthreads();
  }

  const int c0 = colBlock + wCol + l16;
  const int r0 = rowBlock + wRow + 8 * lh;
#pragma unroll
  for (int i = 0; i < 2; i++)
#pragma unroll
    for (int j = 0; j < 4; j++)
      store_tile(acc[i][j], r0 + i * 16, c0 + j * 16, M, bias, outH, outF, mode);
}

// ---- fused attention: one block (4 waves) per (b,h) -----------------------
// scores = (Q K^T)/8, softmax over t, O = P V. S=64, DH=64.

#define QKVLD 1536
#define ALD 72   // f16 row stride (halves)
#define SLD 68   // f32 score row stride

__global__ __launch_bounds__(128) void attn_kernel(
    const _Float16* __restrict__ QKV, _Float16* __restrict__ O) {
  __shared__ __align__(16) _Float16 Qs[S_ * ALD];
  __shared__ __align__(16) _Float16 Ks[S_ * ALD];
  __shared__ __align__(16) _Float16 Vt[DH_ * ALD];   // V transposed [dh][t]
  __shared__ __align__(16) float    Sc[S_ * SLD];
  __shared__ __align__(16) _Float16 Ps[S_ * ALD];

  const int bh = blockIdx.x;
  const int b  = bh >> 3;
  const int h  = bh & 7;
  const int tid  = threadIdx.x;
  const int wave = tid >> 5;
  const int lane = tid & 31;
  const int l16  = lane & 15;
  const int lh   = lane >> 4;

  const int qc = h * DH_;
  const int kc = D_ + h * DH_;
  const int vc = 2 * D_ + h * DH_;

  // load Q, K (row-major) and V transposed into LDS
  {
    const int s   = tid >> 1;          // 0..63
    const int off = (tid & 1) * 32;
    const _Float16* base = QKV + (size_t)(s * B_ + b) * QKVLD;
#pragma unroll
    for (int u = 0; u < 4; u++) {
      *(v8h*)&Qs[s * ALD + off + u * 8] = *(const v8h*)(base + qc + off + u * 8);
      *(v8h*)&Ks[s * ALD + off + u * 8] = *(const v8h*)(base + kc + off + u * 8);
    }
#pragma unroll
    for (int u = 0; u < 32; u++) {
      Vt[(off + u) * ALD + s] = base[vc + off + u];
    }
  }
  __syncthreads();

  // scores: wave w computes rows m0..m0+15, all 64 cols
  const int m0 = wave * 16;
#pragma unroll
  for (int nt = 0; nt < 4; nt++) {
    v8f acc = {};
#pragma unroll
    for (int kk = 0; kk < DH_; kk += 32) {
      v16h a  = load_frag_a(&Qs[m0 * ALD + kk], ALD);
      v16h bf = load_frag_b(&Ks[(nt * 16) * ALD + kk], ALD);
      acc = wmma16x16x32(a, bf, acc);
    }
    const int col = nt * 16 + l16;
#pragma unroll
    for (int v = 0; v < 8; v++)
      Sc[(m0 + 8 * lh + v) * SLD + col] = acc[v] * 0.125f;
  }
  __syncthreads();

  // softmax per row (threads 0..63)
  if (tid < S_) {
    float* r = &Sc[tid * SLD];
    float mx = -3.4e38f;
#pragma unroll 8
    for (int j = 0; j < S_; j++) mx = fmaxf(mx, r[j]);
    float sum = 0.0f;
#pragma unroll 8
    for (int j = 0; j < S_; j++) { float e = __expf(r[j] - mx); r[j] = e; sum += e; }
    const float inv = 1.0f / sum;
    _Float16* p = &Ps[tid * ALD];
#pragma unroll 8
    for (int j = 0; j < S_; j++) p[j] = (_Float16)(r[j] * inv);
  }
  __syncthreads();

  // O = P V : wave w computes rows m0..m0+15, all 64 dh cols
#pragma unroll
  for (int nt = 0; nt < 4; nt++) {
    v8f acc = {};
#pragma unroll
    for (int kk = 0; kk < S_; kk += 32) {
      v16h a  = load_frag_a(&Ps[m0 * ALD + kk], ALD);
      v16h bf = load_frag_b(&Vt[(nt * 16) * ALD + kk], ALD);
      acc = wmma16x16x32(a, bf, acc);
    }
    const int dh = nt * 16 + l16;
#pragma unroll
    for (int v = 0; v < 8; v++) {
      int s = m0 + 8 * lh + v;
      O[(size_t)(s * B_ + b) * D_ + h * DH_ + dh] = (_Float16)acc[v];
    }
  }
}

// ---- pooling + hashing head ----------------------------------------------

__global__ __launch_bounds__(256) void pool_kernel(
    const float* __restrict__ X, float* __restrict__ pooled) {
  size_t idx = (size_t)blockIdx.x * 256 + threadIdx.x;  // over B*D
  int b = (int)(idx >> 9);
  int d = (int)(idx & (D_ - 1));
  float s = 0.0f;
#pragma unroll 8
  for (int t = 0; t < S_; t++)
    s += X[((size_t)t * B_ + b) * D_ + d];
  pooled[idx] = s * (1.0f / S_);
}

__global__ __launch_bounds__(256) void hash_kernel(
    const float* __restrict__ pooled, const float* __restrict__ hw,
    const float* __restrict__ hb, float* __restrict__ out) {
  int idx = blockIdx.x * 256 + threadIdx.x;   // over B*KBITS
  int b = idx >> 5;
  int k = idx & 31;
  const float* pr = pooled + (size_t)b * D_;
  const float* wr = hw + (size_t)k * D_;
  float s = 0.0f;
#pragma unroll 8
  for (int d = 0; d < D_; d++) s += pr[d] * wr[d];
  out[idx] = tanhf(s + hb[k]);
}

// ---- host-side launch sequence -------------------------------------------

extern "C" void kernel_launch(void* const* d_in, const int* in_sizes, int n_in,
                              void* d_out, int out_size, void* d_ws, size_t ws_size,
                              hipStream_t stream) {
  (void)in_sizes; (void)n_in; (void)out_size; (void)ws_size;

  const float* x      = (const float*)d_in[0];
  const float* ln1_g  = (const float*)d_in[1];
  const float* ln1_b  = (const float*)d_in[2];
  const float* qkv_w  = (const float*)d_in[3];
  const float* qkv_b  = (const float*)d_in[4];
  const float* out_w  = (const float*)d_in[5];
  const float* out_b  = (const float*)d_in[6];
  const float* ln2_g  = (const float*)d_in[7];
  const float* ln2_b  = (const float*)d_in[8];
  const float* mlp_w1 = (const float*)d_in[9];
  const float* mlp_b1 = (const float*)d_in[10];
  const float* mlp_w2 = (const float*)d_in[11];
  const float* mlp_b2 = (const float*)d_in[12];
  const float* hash_w = (const float*)d_in[13];
  const float* hash_b = (const float*)d_in[14];
  float* out = (float*)d_out;

  char* ws = (char*)d_ws;
  float*     X      = (float*)ws;                                   // 128 MB
  _Float16*  Hbuf   = (_Float16*)(ws + ((size_t)128 << 20));        //  64 MB
  _Float16*  QKVb   = (_Float16*)(ws + ((size_t)192 << 20));        // 192 MB
  _Float16*  Obuf   = (_Float16*)(ws + ((size_t)384 << 20));        //  64 MB
  _Float16*  H1     = (_Float16*)(ws + ((size_t)448 << 20));        // 256 MB
  _Float16*  Wf16   = (_Float16*)(ws + ((size_t)704 << 20));        //   4 MB
  float*     pooled = (float*)(ws + ((size_t)708 << 20));           //   2 MB

  pe_add_kernel<<<(NTOK * D_) / 256, 256, 0, stream>>>(x, X);

  for (int l = 0; l < NL_; l++) {
    // --- attention block ---
    ln_kernel<<<NTOK / 8, 256, 0, stream>>>(X, ln1_g + l * D_, ln1_b + l * D_, Hbuf);
    f2h_kernel<<<(3 * D_ * D_) / 256, 256, 0, stream>>>(qkv_w + (size_t)l * 3 * D_ * D_, Wf16);
    gemm_f16_kernel<<<dim3((3 * D_) / BN, NTOK / BM), 256, 0, stream>>>(
        Hbuf, Wf16, qkv_b + l * 3 * D_, QKVb, nullptr, 3 * D_, D_, 0);
    attn_kernel<<<B_ * H_, 128, 0, stream>>>(QKVb, Obuf);
    f2h_kernel<<<(D_ * D_) / 256, 256, 0, stream>>>(out_w + (size_t)l * D_ * D_, Wf16);
    gemm_f16_kernel<<<dim3(D_ / BN, NTOK / BM), 256, 0, stream>>>(
        Obuf, Wf16, out_b + l * D_, nullptr, X, D_, D_, 1);

    // --- MLP block ---
    ln_kernel<<<NTOK / 8, 256, 0, stream>>>(X, ln2_g + l * D_, ln2_b + l * D_, Hbuf);
    f2h_kernel<<<(DFF_ * D_) / 256, 256, 0, stream>>>(mlp_w1 + (size_t)l * DFF_ * D_, Wf16);
    gemm_f16_kernel<<<dim3(DFF_ / BN, NTOK / BM), 256, 0, stream>>>(
        Hbuf, Wf16, mlp_b1 + l * DFF_, H1, nullptr, DFF_, D_, 2);
    f2h_kernel<<<(D_ * DFF_) / 256, 256, 0, stream>>>(mlp_w2 + (size_t)l * D_ * DFF_, Wf16);
    gemm_f16_kernel<<<dim3(D_ / BN, NTOK / BM), 256, 0, stream>>>(
        H1, Wf16, mlp_b2 + l * D_, nullptr, X, D_, DFF_, 1);
  }

  pool_kernel<<<(B_ * D_) / 256, 256, 0, stream>>>(X, pooled);
  hash_kernel<<<(B_ * KBITS) / 256, 256, 0, stream>>>(pooled, hash_w, hash_b, out);
}
